// Seq2Seq_48481590837544
// MI455X (gfx1250) — matrix operand order repaired and checked by reading, not compiled, not run on hardware
//
#include <hip/hip_runtime.h>

// ---------------- problem constants ----------------
#define BATCH   256
#define TSTEPS  128
#define INDIM   64
#define HD      1024
#define NL      3
#define OUTD    2
#define GATES_N 4096          // 4*HD
#define BH      (BATCH * HD)  // 262144
#define DEC_K   1056          // HD+OUTD=1026 padded up to multiple of 32

// ---------------- launch config ----------------
#define BDIM 256
#define GBLK 128              // persistent blocks (must be co-resident)

typedef float          v8f     __attribute__((ext_vector_type(8)));
typedef __bf16         bf16x16 __attribute__((ext_vector_type(16)));
typedef unsigned int   u32x4   __attribute__((ext_vector_type(4)));
typedef unsigned short u16;

union ABFrag { u32x4 q[2]; bf16x16 v; };

// ---------------- small helpers ----------------
__device__ __forceinline__ u16 f2bf(float f) {
  unsigned u = __float_as_uint(f);
  u += 0x7FFFu + ((u >> 16) & 1u);       // round-to-nearest-even
  return (u16)(u >> 16);
}
__device__ __forceinline__ float bf2f(u16 h) {
  return __uint_as_float(((unsigned)h) << 16);
}
__device__ __forceinline__ float sigm(float x)  { return 1.0f / (1.0f + __expf(-x)); }
__device__ __forceinline__ float tanhx(float x) { return 2.0f / (1.0f + __expf(-2.0f * x)) - 1.0f; }

// Sense-reversing grid barrier (all GBLK blocks resident).
__device__ __forceinline__ void gridbar(unsigned* bar) {
  __syncthreads();
  if (threadIdx.x == 0) {
    volatile unsigned* vb = (volatile unsigned*)bar;
    unsigned gen = vb[1];
    __threadfence();
    if (atomicAdd(&bar[0], 1u) == gridDim.x - 1) {
      vb[0] = 0;
      __threadfence();
      atomicAdd(&bar[1], 1u);
    } else {
      while (vb[1] == gen) { __builtin_amdgcn_s_sleep(2); }
    }
    __threadfence();
  }
  __syncthreads();
}

// ---------------- WMMA gate GEMM ----------------
// 2x2 register-blocked 32x32 macro-tile per wave: 4 independent accumulator
// chains keep the matrix pipe busy; 8 b128 loads feed 4 WMMAs per k-chunk.
// CDNA5 16-bit fragment layout:
//   A: lanes 0-15 hold M=lane, K chunks {0-7,16-23}; lanes 16-31 hold {8-15,24-31}
//   B: lanes 0-15 hold N=lane, K chunks {0-7,16-23}; lanes 16-31 hold {8-15,24-31}
__device__ __forceinline__ v8f wmma_bf16(const ABFrag& a, const ABFrag& b, v8f c) {
  return __builtin_amdgcn_wmma_f32_16x16x32_bf16(false, a.v, false, b.v,
                                                 (short)0, c, false, false);
}

__device__ __forceinline__ void wmma_block22(const u16* __restrict__ A, int lda,
                                             const u16* __restrict__ B, int ldb, int K,
                                             int r, int hi,
                                             v8f& c00, v8f& c01, v8f& c10, v8f& c11) {
  const u16* a0 = A + (size_t)r * lda;
  const u16* a1 = A + (size_t)(16 + r) * lda;
  const u16* b0 = B + (size_t)r * ldb;
  const u16* b1 = B + (size_t)(16 + r) * ldb;
  for (int k0 = 0; k0 < K; k0 += 32) {
    ABFrag A0, A1, B0, B1;
    A0.q[0] = *(const u32x4*)(a0 + k0 + hi * 8);
    A0.q[1] = *(const u32x4*)(a0 + k0 + 16 + hi * 8);
    A1.q[0] = *(const u32x4*)(a1 + k0 + hi * 8);
    A1.q[1] = *(const u32x4*)(a1 + k0 + 16 + hi * 8);
    B0.q[0] = *(const u32x4*)(b0 + k0 + hi * 8);
    B0.q[1] = *(const u32x4*)(b0 + k0 + 16 + hi * 8);
    B1.q[0] = *(const u32x4*)(b1 + k0 + hi * 8);
    B1.q[1] = *(const u32x4*)(b1 + k0 + 16 + hi * 8);
    c00 = wmma_bf16(A0, B0, c00);
    c01 = wmma_bf16(A0, B1, c01);
    c10 = wmma_bf16(A1, B0, c10);
    c11 = wmma_bf16(A1, B1, c11);
  }
}

__device__ __forceinline__ void store_tile(float* __restrict__ gates,
                                           int m0, int n0, int r, int hi, v8f acc) {
  float* g = gates + (size_t)(m0 + hi * 8) * GATES_N + n0 + r;
#pragma unroll
  for (int j = 0; j < 8; ++j) g[(size_t)j * GATES_N] = acc[j];
}

// gates[256,4096] = bias + X[256,Kin]*Wih^T + H[256,1024]*Whh^T
// Block patch layout: 4 m-patches x 32 n-patches; 8 waves/block tile a 2x4
// sub-grid of 32x32 macro-tiles so A/B streams are WGP$-shared.
__device__ __forceinline__ void gemm_gates(const u16* __restrict__ X, int lda, int Kin,
                                           const u16* __restrict__ Wih,
                                           const u16* __restrict__ Whh,
                                           const u16* __restrict__ Hst,
                                           const float* __restrict__ bias,
                                           float* __restrict__ gates) {
  const int wib  = threadIdx.x >> 5;            // wave in block: 0..7
  const int bm   = blockIdx.x & 3;              // 4 m patches (64 rows each)
  const int bn   = blockIdx.x >> 2;             // 32 n patches (128 cols each)
  const int m0   = (((bm << 1) | (wib & 1)) << 5);    // 0..224, step 32
  const int n0   = (((bn << 2) | (wib >> 1)) << 5);   // 0..4064, step 32
  const int lane = threadIdx.x & 31;
  const int r    = lane & 15;
  const int hi   = lane >> 4;

  const float bv0 = bias[n0 + r];
  const float bv1 = bias[n0 + 16 + r];
  v8f c00 = {bv0, bv0, bv0, bv0, bv0, bv0, bv0, bv0};
  v8f c01 = {bv1, bv1, bv1, bv1, bv1, bv1, bv1, bv1};
  v8f c10 = c00, c11 = c01;

  wmma_block22(X   + (size_t)m0 * lda, lda, Wih + (size_t)n0 * Kin, Kin, Kin,
               r, hi, c00, c01, c10, c11);
  wmma_block22(Hst + (size_t)m0 * HD,  HD,  Whh + (size_t)n0 * HD,  HD,  HD,
               r, hi, c00, c01, c10, c11);

  store_tile(gates, m0,      n0,      r, hi, c00);
  store_tile(gates, m0,      n0 + 16, r, hi, c01);
  store_tile(gates, m0 + 16, n0,      r, hi, c10);
  store_tile(gates, m0 + 16, n0 + 16, r, hi, c11);
}

// Elementwise LSTM cell update: i,f,g,o gate order.
__device__ __forceinline__ void lstm_update(const float* __restrict__ gates,
                                            float* __restrict__ c,
                                            u16* __restrict__ h) {
  const int gtid = blockIdx.x * blockDim.x + threadIdx.x;
  const int nth  = gridDim.x * blockDim.x;
  for (int i = gtid; i < BH; i += nth) {
    const int b = i >> 10, j = i & (HD - 1);
    const float* g = gates + (size_t)b * GATES_N;
    const float ig = sigm(g[j]);
    const float fg = sigm(g[HD + j]);
    const float gg = tanhx(g[2 * HD + j]);
    const float og = sigm(g[3 * HD + j]);
    const float cc = fg * c[i] + ig * gg;
    c[i] = cc;
    h[i] = f2bf(og * tanhx(cc));
  }
}

// ---------------- persistent encoder ----------------
__global__ __launch_bounds__(BDIM) void enc_kernel(
    const u16* __restrict__ x_bf,          // [B,T,IN] bf16
    const u16* __restrict__ Wih0,          // [4096,64]
    const u16* __restrict__ WihR,          // [2,4096,1024]
    const u16* __restrict__ Whh,           // [3,4096,1024]
    const float* __restrict__ bias,        // [3,4096] (bih+bhh)
    u16* __restrict__ h, float* __restrict__ c,   // [3,B,H]
    float* __restrict__ gates, unsigned* bar) {
  const int gtid = blockIdx.x * blockDim.x + threadIdx.x;
  const int nth  = gridDim.x * blockDim.x;
  for (int i = gtid; i < NL * BH; i += nth) { h[i] = 0; c[i] = 0.0f; }
  gridbar(bar);
  for (int t = 0; t < TSTEPS; ++t) {
    for (int l = 0; l < NL; ++l) {
      const u16* X; int lda, K; const u16* Wih;
      if (l == 0) { X = x_bf + t * INDIM; lda = TSTEPS * INDIM; K = INDIM; Wih = Wih0; }
      else        { X = h + (l - 1) * BH; lda = HD; K = HD;
                    Wih = WihR + (size_t)(l - 1) * GATES_N * HD; }
      gemm_gates(X, lda, K, Wih, Whh + (size_t)l * GATES_N * HD,
                 h + l * BH, bias + l * GATES_N, gates);
      gridbar(bar);
      lstm_update(gates, c + l * BH, h + l * BH);
      gridbar(bar);
    }
  }
}

// ---------------- persistent decoder ----------------
__global__ __launch_bounds__(BDIM) void dec_kernel(
    const u16* __restrict__ Wih0,          // [4096,1056] (padded)
    const u16* __restrict__ WihR,          // [2,4096,1024]
    const u16* __restrict__ Whh,           // [3,4096,1024]
    const float* __restrict__ bias,        // [3,4096]
    const u16* __restrict__ fcW,           // [2,1024] bf16
    const float* __restrict__ fcb,         // [2]
    const int* __restrict__ tsl_p,
    u16* __restrict__ h, float* __restrict__ c,   // [3,B,H] (carry encoder finals)
    u16* __restrict__ dec_in,              // [B,1056] bf16 (pre-zeroed)
    float* __restrict__ gates,
    float* __restrict__ out,               // [B,tsl,2] f32
    unsigned* bar) {
  const int tsl  = *tsl_p;
  const int gtid = blockIdx.x * blockDim.x + threadIdx.x;
  const int nth  = gridDim.x * blockDim.x;
  const int wave = gtid >> 5;
  const int lane = threadIdx.x & 31;
  for (int t = 0; t < tsl; ++t) {
    for (int l = 0; l < NL; ++l) {
      const u16* X; int lda, K; const u16* Wih;
      if (l == 0) { X = dec_in; lda = DEC_K; K = DEC_K; Wih = Wih0; }
      else        { X = h + (l - 1) * BH; lda = HD; K = HD;
                    Wih = WihR + (size_t)(l - 1) * GATES_N * HD; }
      gemm_gates(X, lda, K, Wih, Whh + (size_t)l * GATES_N * HD,
                 h + l * BH, bias + l * GATES_N, gates);
      gridbar(bar);
      lstm_update(gates, c + l * BH, h + l * BH);
      gridbar(bar);
    }
    // FC head + feedback input concat(h_top, out)
    const u16* h2 = h + 2 * BH;
    for (int i = gtid; i < BH; i += nth) {
      const int b = i >> 10, j = i & (HD - 1);
      dec_in[b * DEC_K + j] = h2[i];
    }
    if (wave < BATCH * OUTD) {               // one wave per (b, o) dot product
      const int b = wave >> 1, o = wave & 1;
      const u16* hr = h2 + (size_t)b * HD;
      const u16* wr = fcW + (size_t)o * HD;
      float s = 0.0f;
      const int k0 = lane * (HD / 32);
      for (int k = k0; k < k0 + HD / 32; ++k) s += bf2f(hr[k]) * bf2f(wr[k]);
      for (int off = 16; off; off >>= 1) s += __shfl_xor(s, off, 32);
      if (lane == 0) {
        const float ov = s + fcb[o];
        out[((size_t)b * tsl + t) * OUTD + o] = ov;
        dec_in[b * DEC_K + HD + o] = f2bf(ov);
      }
    }
    gridbar(bar);
  }
}

// ---------------- prep kernels ----------------
__global__ void k_f32_to_bf16(const float* __restrict__ s, u16* __restrict__ d, int n) {
  int i = blockIdx.x * blockDim.x + threadIdx.x, st = gridDim.x * blockDim.x;
  for (; i < n; i += st) d[i] = f2bf(s[i]);
}
__global__ void k_f32_to_bf16_pad(const float* __restrict__ s, u16* __restrict__ d,
                                  int rows, int scols, int dcols) {
  int i = blockIdx.x * blockDim.x + threadIdx.x, st = gridDim.x * blockDim.x;
  const int n = rows * dcols;
  for (; i < n; i += st) {
    const int r = i / dcols, cc = i - r * dcols;
    d[i] = (cc < scols) ? f2bf(s[(size_t)r * scols + cc]) : (u16)0;
  }
}
__global__ void k_add_bias(const float* __restrict__ a, const float* __restrict__ b,
                           float* __restrict__ o, int n) {
  int i = blockIdx.x * blockDim.x + threadIdx.x, st = gridDim.x * blockDim.x;
  for (; i < n; i += st) o[i] = a[i] + b[i];
}
__global__ void k_zero_u16(u16* __restrict__ p, int n) {
  int i = blockIdx.x * blockDim.x + threadIdx.x, st = gridDim.x * blockDim.x;
  for (; i < n; i += st) p[i] = 0;
}
__global__ void k_init_bar(unsigned* bar) { bar[0] = 0u; bar[1] = 0u; }

// ---------------- host launcher ----------------
extern "C" void kernel_launch(void* const* d_in, const int* in_sizes, int n_in,
                              void* d_out, int out_size, void* d_ws, size_t ws_size,
                              hipStream_t stream) {
  (void)in_sizes; (void)n_in; (void)out_size; (void)ws_size;
  const float* x         = (const float*)d_in[0];
  const float* eWih0f    = (const float*)d_in[1];
  const float* eWihRf    = (const float*)d_in[2];
  const float* eWhhf     = (const float*)d_in[3];
  const float* ebih      = (const float*)d_in[4];
  const float* ebhh      = (const float*)d_in[5];
  const float* dWih0f    = (const float*)d_in[6];
  const float* dWihRf    = (const float*)d_in[7];
  const float* dWhhf     = (const float*)d_in[8];
  const float* dbih      = (const float*)d_in[9];
  const float* dbhh      = (const float*)d_in[10];
  const float* fcWf      = (const float*)d_in[11];
  const float* fcb       = (const float*)d_in[12];
  const int*   tsl       = (const int*)d_in[13];
  float* out = (float*)d_out;

  // workspace carve-up (256B aligned)
  unsigned char* ws = (unsigned char*)d_ws;
  size_t off = 0;
  auto alloc = [&](size_t bytes) -> void* {
    void* p = ws + off;
    off = (off + bytes + 255) & ~(size_t)255;
    return p;
  };
  u16*   x_bf   = (u16*)  alloc((size_t)BATCH * TSTEPS * INDIM * 2);
  u16*   eWih0  = (u16*)  alloc((size_t)GATES_N * INDIM * 2);
  u16*   eWihR  = (u16*)  alloc((size_t)2 * GATES_N * HD * 2);
  u16*   eWhh   = (u16*)  alloc((size_t)3 * GATES_N * HD * 2);
  u16*   dWih0  = (u16*)  alloc((size_t)GATES_N * DEC_K * 2);
  u16*   dWihR  = (u16*)  alloc((size_t)2 * GATES_N * HD * 2);
  u16*   dWhh   = (u16*)  alloc((size_t)3 * GATES_N * HD * 2);
  u16*   fcWbf  = (u16*)  alloc((size_t)OUTD * HD * 2);
  float* ebias  = (float*)alloc((size_t)3 * GATES_N * 4);
  float* dbias  = (float*)alloc((size_t)3 * GATES_N * 4);
  float* gates  = (float*)alloc((size_t)BATCH * GATES_N * 4);
  u16*   hstate = (u16*)  alloc((size_t)NL * BH * 2);
  float* cstate = (float*)alloc((size_t)NL * BH * 4);
  u16*   dec_in = (u16*)  alloc((size_t)BATCH * DEC_K * 2);
  unsigned* bar = (unsigned*)alloc(256);

  const int CB = 512, CT = 256;  // conversion grid
  k_f32_to_bf16<<<CB, CT, 0, stream>>>(x,      x_bf,  BATCH * TSTEPS * INDIM);
  k_f32_to_bf16<<<CB, CT, 0, stream>>>(eWih0f, eWih0, GATES_N * INDIM);
  k_f32_to_bf16<<<CB, CT, 0, stream>>>(eWihRf, eWihR, 2 * GATES_N * HD);
  k_f32_to_bf16<<<CB, CT, 0, stream>>>(eWhhf,  eWhh,  3 * GATES_N * HD);
  k_f32_to_bf16_pad<<<CB, CT, 0, stream>>>(dWih0f, dWih0, GATES_N, HD + OUTD, DEC_K);
  k_f32_to_bf16<<<CB, CT, 0, stream>>>(dWihRf, dWihR, 2 * GATES_N * HD);
  k_f32_to_bf16<<<CB, CT, 0, stream>>>(dWhhf,  dWhh,  3 * GATES_N * HD);
  k_f32_to_bf16<<<CB, CT, 0, stream>>>(fcWf,   fcWbf, OUTD * HD);
  k_add_bias<<<CB, CT, 0, stream>>>(ebih, ebhh, ebias, 3 * GATES_N);
  k_add_bias<<<CB, CT, 0, stream>>>(dbih, dbhh, dbias, 3 * GATES_N);
  k_zero_u16<<<CB, CT, 0, stream>>>(dec_in, BATCH * DEC_K);
  k_init_bar<<<1, 1, 0, stream>>>(bar);

  enc_kernel<<<GBLK, BDIM, 0, stream>>>(x_bf, eWih0, eWihR, eWhh, ebias,
                                        hstate, cstate, gates, bar);
  dec_kernel<<<GBLK, BDIM, 0, stream>>>(dWih0, dWihR, dWhh, dbias, fcWbf, fcb, tsl,
                                        hstate, cstate, dec_in, gates, out, bar);
}
